// MSAColumnAttention_9268539425319
// MI455X (gfx1250) — matrix operand order repaired and checked by reading, not compiled, not run on hardware
//
#include <hip/hip_runtime.h>
#include <hip/hip_bf16.h>
#include <math.h>

// ---------------- CDNA5 WMMA types/helpers (wave32) ----------------
typedef _Float16 f16x16 __attribute__((ext_vector_type(16)));
typedef _Float16 f16x8  __attribute__((ext_vector_type(8)));
typedef float    f32x8  __attribute__((ext_vector_type(8)));
typedef unsigned int u32x4 __attribute__((ext_vector_type(4)));
typedef int      i32x4 __attribute__((ext_vector_type(4)));
typedef int      i32x8 __attribute__((ext_vector_type(8)));

#define WMMA(a, b, c) \
  __builtin_amdgcn_wmma_f32_16x16x32_f16(false, (a), false, (b), (short)0, (c), false, false)

// A-fragment for 16x16x32 f16: lane holds 16 halves in two contiguous runs of 8.
static __device__ __forceinline__ f16x16 load_a_split(const _Float16* lo, const _Float16* hi) {
  f16x8 l = *(const f16x8*)lo;
  f16x8 h = *(const f16x8*)hi;
  f16x16 r;
#pragma unroll
  for (int j = 0; j < 8; ++j) { r[j] = l[j]; r[j + 8] = h[j]; }
  return r;
}
// B-fragment: lane holds 16 contiguous halves (32B).
static __device__ __forceinline__ f16x16 load16(const _Float16* p) { return *(const f16x16*)p; }

static __device__ __forceinline__ float red_max16(float v) {
#pragma unroll
  for (int m = 1; m < 16; m <<= 1) v = fmaxf(v, __shfl_xor(v, m, 32));
  return v;
}
static __device__ __forceinline__ float red_sum16(float v) {
#pragma unroll
  for (int m = 1; m < 16; m <<= 1) v += __shfl_xor(v, m, 32);
  return v;
}

// Problem constants
#define SEQ 512
#define RES 128
#define CH  256
#define NH  8
#define DH  32
#define NPOS (SEQ * RES)            // 65536 positions (s,r)
#define QSCALE 0.17677669529663687f // 1/sqrt(32)

// ---------------- Tensor Data Mover: 1-D contiguous copy global -> LDS ----------------
// D# packing per CDNA5 ISA ch.8: group0 = {count=1, lds_addr, global_addr, type=2},
// group1 = {data_size=2B, tensor_dim0=tile_dim0=nelem, stride=nelem}, groups 2/3 zero.
#if __has_builtin(__builtin_amdgcn_tensor_load_to_lds)
#define HAS_TDM 1
static __device__ __forceinline__ void tdm_load_1d_f16(unsigned lds_off, const void* gsrc,
                                                       unsigned nelem /* <= 65535 */) {
  unsigned long long ga = (unsigned long long)gsrc;
  u32x4 g0 = {1u,                                   // count=1 (valid user descriptor)
              lds_off,                              // lds_addr (bytes)
              (unsigned)(ga & 0xffffffffu),         // global_addr[31:0]
              (unsigned)((ga >> 32) & 0x01ffffffu) | (2u << 30)};  // addr[56:32] | type=2
  i32x8 g1 = {(int)0x00010000u,                      // data_size=1 -> 2 bytes; mask=0
              (int)((nelem & 0xffffu) << 16),        // tensor_dim0[15:0] @ bits63:48
              (int)(((nelem >> 16) & 0xffffu) | (1u << 16)),  // dim0 hi | tensor_dim1=1
              (int)((nelem & 0xffffu) << 16),        // tile_dim0 @ bits127:112
              1,                                     // tile_dim1=1
              (int)nelem,                            // tensor_dim0_stride lo32
              0, 0};
  i32x4 gz = {0, 0, 0, 0};
#if defined(__clang_major__) && (__clang_major__ >= 23)
  i32x8 g4 = {0, 0, 0, 0, 0, 0, 0, 0};
  __builtin_amdgcn_tensor_load_to_lds(g0, g1, gz, gz, g4, 0);
#else
  __builtin_amdgcn_tensor_load_to_lds(g0, g1, gz, gz, 0);
#endif
}
#else
#define HAS_TDM 0
#endif

// ---------------- Kernel 0: pack weights to f16 B-fragment layout ----------------
__global__ void msa_pack_w(const float* __restrict__ wq, const float* __restrict__ wk,
                           const float* __restrict__ wv, const float* __restrict__ wg,
                           const float* __restrict__ wf, _Float16* __restrict__ wpk) {
  int t = blockIdx.x * 256 + threadIdx.x;     // 80*256 = 20480 threads
  int lane = t & 31;
  int ks = (t >> 5) & 7;
  int nt = t >> 8;                            // 0..79
  const float* src;
  int col;
  if (nt < 64) {
    int mat = nt >> 4;
    src = (mat == 0) ? wq : (mat == 1) ? wk : (mat == 2) ? wv : wg;
    col = ((nt & 15) << 4) + (lane & 15);
  } else {
    src = wf;
    col = ((nt - 64) << 4) + (lane & 15);
  }
  int krow = ks * 32 + ((lane >= 16) ? 16 : 0);
  _Float16* dst = wpk + (((size_t)(nt * 8 + ks) * 32 + lane) << 4);
#pragma unroll
  for (int j = 0; j < 16; ++j) dst[j] = (_Float16)src[(size_t)(krow + j) * CH + col];
}

// ---------------- Kernel 1: LayerNorm + fused Q/K/V/Gate projection ----------------
// Block = 256 threads = 8 waves, 16 positions per block. GEMM [16 x 256] x [256 x 1024].
__global__ void msa_ln_qkvg(const float* __restrict__ x, const float* __restrict__ lnw,
                            const float* __restrict__ lnb, const float* __restrict__ bg,
                            const _Float16* __restrict__ wpk, _Float16* __restrict__ qb,
                            _Float16* __restrict__ kb, _Float16* __restrict__ vb,
                            _Float16* __restrict__ gb) {
  __shared__ _Float16 xs[16 * 264];

  const int tid  = threadIdx.x;
  const int lane = tid & 31;
  const int wv   = tid >> 5;
  const int hl   = (lane >> 4) & 1;
  const int p0   = blockIdx.x << 4;

  // ---- LayerNorm: 16 threads per position, 16 floats each ----
  {
    const int pos = tid >> 4;
    const int l16 = tid & 15;
    const float4* x4 = (const float4*)(x + (size_t)(p0 + pos) * CH + (l16 << 4));
    float4 a = x4[0], b = x4[1], c = x4[2], d = x4[3];
    float vals[16] = {a.x, a.y, a.z, a.w, b.x, b.y, b.z, b.w,
                      c.x, c.y, c.z, c.w, d.x, d.y, d.z, d.w};
    float s = 0.f, ss = 0.f;
#pragma unroll
    for (int j = 0; j < 16; ++j) { s += vals[j]; ss += vals[j] * vals[j]; }
    s  = red_sum16(s);
    ss = red_sum16(ss);
    float mu   = s * (1.0f / CH);
    float var  = ss * (1.0f / CH) - mu * mu;
    float rstd = rsqrtf(var + 1e-5f);
#pragma unroll
    for (int j = 0; j < 16; ++j) {
      int c_ = (l16 << 4) + j;
      xs[pos * 264 + c_] = (_Float16)((vals[j] - mu) * rstd * lnw[c_] + lnb[c_]);
    }
  }
  __syncthreads();

  // ---- A fragments (shared by all 8 N-tiles of this wave) ----
  f16x16 afr[8];
#pragma unroll
  for (int ks = 0; ks < 8; ++ks) {
    const _Float16* arow = &xs[(lane & 15) * 264 + ks * 32 + 16 * hl];
    afr[ks] = load_a_split(arow, arow + 8);
  }

  // ---- WMMA GEMM, fully unrolled so `mat` is compile-time (no branchy epilogue) ----
#pragma unroll
  for (int nt8 = 0; nt8 < 8; ++nt8) {
    const int nt  = wv + (nt8 << 3);
    const int mat = nt8 >> 1;  // == nt >> 4 for wv in [0,8): compile-time constant
    const _Float16* bbase = wpk + (((size_t)nt * 8 * 32 + lane) << 4);
    f16x16 bfr[8];
#pragma unroll
    for (int ks = 0; ks < 8; ++ks) bfr[ks] = load16(bbase + (size_t)ks * 32 * 16);
    f32x8 acc = {0.f, 0.f, 0.f, 0.f, 0.f, 0.f, 0.f, 0.f};
#pragma unroll
    for (int ks = 0; ks < 8; ++ks) acc = WMMA(afr[ks], bfr[ks], acc);

    const int col = ((nt << 4) + (lane & 15)) & 255;
    const int hh  = col >> 5;
    const int dh  = col & 31;
#pragma unroll
    for (int i = 0; i < 8; ++i) {
      int p = p0 + i + 8 * hl;          // position = s*RES + r
      int s_ = p >> 7;
      int r_ = p & 127;
      size_t dst = ((size_t)(r_ * NH + hh) * SEQ + s_) * DH + dh;
      float v = acc[i];
      if (mat == 0)      qb[dst] = (_Float16)(v * QSCALE);
      else if (mat == 1) kb[dst] = (_Float16)v;
      else if (mat == 2) vb[dst] = (_Float16)v;
      else {
        float g = v + bg[col];
        gb[dst] = (_Float16)(1.0f / (1.0f + __expf(-g)));
      }
    }
  }
}

// ---------------- Kernel 2: flash attention over the MSA (S) dimension ----------------
// Block = 8 waves, one (query-tile of 128, r, h). Wave owns 16 query rows.
// Dynamic LDS layout: [ V row-major 32768B | V^T 33280B | P scratch 10240B ]
#define VTSTR 520
#define PSTR  40
#define LDS_VRM 0
#define LDS_VT  32768
#define LDS_PLS (32768 + 33280)
#define LDS_TOTAL (32768 + 33280 + 10240)
__global__ void msa_attn(const _Float16* __restrict__ qbuf, const _Float16* __restrict__ kbuf,
                         const _Float16* __restrict__ vbuf, const _Float16* __restrict__ gbuf,
                         _Float16* __restrict__ obuf) {
  extern __shared__ char smem[];
  _Float16* vrm = (_Float16*)(smem + LDS_VRM);
  _Float16* vT  = (_Float16*)(smem + LDS_VT);
  _Float16* pls = (_Float16*)(smem + LDS_PLS);

  const int tid  = threadIdx.x;
  const int lane = tid & 31;
  const int wv   = tid >> 5;
  const int hl   = (lane >> 4) & 1;
  const int bx   = blockIdx.x;             // 4 * 128 * 8 = 4096 blocks
  const int h    = bx & 7;
  const int r    = (bx >> 3) & 127;
  const int qt   = bx >> 10;
  const int q0   = qt * 128 + wv * 16;

  const size_t base = (size_t)(r * NH + h) * SEQ * DH;
  const _Float16* qp = qbuf + base;
  const _Float16* kp = kbuf + base;
  const _Float16* vp = vbuf + base;
  const _Float16* gp = gbuf + base;
  _Float16* op = obuf + base;

  // ---- Stage V into LDS: TDM DMA (async tensor path), then transpose in LDS ----
#if HAS_TDM
  if (wv == 0) {
    tdm_load_1d_f16(LDS_VRM, vp, SEQ * DH);    // 32KB contiguous, TENSORcnt-tracked
    __builtin_amdgcn_s_wait_tensorcnt(0);
  }
#else
  for (int e = tid; e < SEQ * DH; e += 256) vrm[e] = vp[e];
#endif
  __syncthreads();
  for (int e = tid; e < SEQ * DH; e += 256) {
    vT[(e & 31) * VTSTR + (e >> 5)] = vrm[e];
  }
  __syncthreads();

  // Q A-fragment for this wave's 16 rows (held for the whole loop).
  const _Float16* qrow = qp + (size_t)(q0 + (lane & 15)) * DH + 16 * hl;
  f16x16 aq = load_a_split(qrow, qrow + 8);

  float rm[8], ls[8];
  f32x8 o0 = {0.f, 0.f, 0.f, 0.f, 0.f, 0.f, 0.f, 0.f};
  f32x8 o1 = o0;
#pragma unroll
  for (int i = 0; i < 8; ++i) { rm[i] = -1e30f; ls[i] = 0.f; }

  _Float16* pl = &pls[wv * 16 * PSTR];

  for (int t = 0; t < SEQ / 32; ++t) {
    const int keyb = t * 32;
    // Scores: Q (16x32) @ K^T -> two 16x16 tiles; K fragments contiguous from global.
    f16x16 bk0 = load16(kp + (size_t)(keyb + (lane & 15)) * DH + 16 * hl);
    f16x16 bk1 = load16(kp + (size_t)(keyb + 16 + (lane & 15)) * DH + 16 * hl);
    f32x8 zz = {0.f, 0.f, 0.f, 0.f, 0.f, 0.f, 0.f, 0.f};
    f32x8 s0 = WMMA(aq, bk0, zz);
    f32x8 s1 = WMMA(aq, bk1, zz);

    // Online softmax update (per accumulator row).
#pragma unroll
    for (int i = 0; i < 8; ++i) {
      float mx   = red_max16(fmaxf(s0[i], s1[i]));
      float nmax = fmaxf(rm[i], mx);
      float corr = __expf(rm[i] - nmax);
      float p0   = __expf(s0[i] - nmax);
      float p1   = __expf(s1[i] - nmax);
      ls[i] = ls[i] * corr + red_sum16(p0 + p1);
      rm[i] = nmax;
      o0[i] *= corr;
      o1[i] *= corr;
      int row = i + 8 * hl;
      pl[row * PSTR + (lane & 15)]      = (_Float16)p0;
      pl[row * PSTR + 16 + (lane & 15)] = (_Float16)p1;
    }
    // P C-layout -> A-fragment via per-wave LDS scratch (in-order within a wave,
    // no cross-wave sharing: no block barrier needed).
    const _Float16* prow = pl + (lane & 15) * PSTR + 16 * hl;
    f16x16 pa = load_a_split(prow, prow + 8);
    f16x16 bv0 = load16(&vT[(lane & 15) * VTSTR + keyb + 16 * hl]);
    f16x16 bv1 = load16(&vT[(16 + (lane & 15)) * VTSTR + keyb + 16 * hl]);
    o0 = WMMA(pa, bv0, o0);
    o1 = WMMA(pa, bv1, o1);
  }

  // Normalize, gate, store f16.
#pragma unroll
  for (int i = 0; i < 8; ++i) {
    float inv = 1.0f / ls[i];
    int row = q0 + i + 8 * hl;
    float g0 = (float)gp[(size_t)row * DH + (lane & 15)];
    float g1 = (float)gp[(size_t)row * DH + 16 + (lane & 15)];
    op[(size_t)row * DH + (lane & 15)]      = (_Float16)(o0[i] * inv * g0);
    op[(size_t)row * DH + 16 + (lane & 15)] = (_Float16)(o1[i] * inv * g1);
  }
}

// ---------------- Kernel 3: output projection o @ wf + bf -> [S,R,C] fp32 ----------------
__global__ void msa_outproj(const _Float16* __restrict__ obuf, const _Float16* __restrict__ wpk,
                            const float* __restrict__ bf, float* __restrict__ out) {
  const int tid  = threadIdx.x;
  const int lane = tid & 31;
  const int wv   = tid >> 5;
  const int hl   = (lane >> 4) & 1;
  const int bx   = blockIdx.x;        // 32 * 128 = 4096 blocks
  const int r    = bx & 127;
  const int s0   = (bx >> 7) << 4;

  const int nt0 = wv, nt1 = wv + 8;   // two of 16 N-tiles per wave
  // Preload all fragments, then run the WMMA chain back-to-back.
  f16x16 afr[8], b0f[8], b1f[8];
#pragma unroll
  for (int ks = 0; ks < 8; ++ks) {    // K-step == head
    const _Float16* abase =
        obuf + ((size_t)(r * NH + ks) * SEQ + s0 + (lane & 15)) * DH + 16 * hl;
    afr[ks] = load_a_split(abase, abase + 8);
    b0f[ks] = load16(wpk + (((size_t)((64 + nt0) * 8 + ks) * 32 + lane) << 4));
    b1f[ks] = load16(wpk + (((size_t)((64 + nt1) * 8 + ks) * 32 + lane) << 4));
  }
  f32x8 acc0 = {0.f, 0.f, 0.f, 0.f, 0.f, 0.f, 0.f, 0.f};
  f32x8 acc1 = acc0;
#pragma unroll
  for (int ks = 0; ks < 8; ++ks) {
    acc0 = WMMA(afr[ks], b0f[ks], acc0);
    acc1 = WMMA(afr[ks], b1f[ks], acc1);
  }
#pragma unroll
  for (int i = 0; i < 8; ++i) {
    int s_ = s0 + i + 8 * hl;
    int c0 = (nt0 << 4) + (lane & 15);
    int c1 = (nt1 << 4) + (lane & 15);
    size_t rowoff = ((size_t)s_ * RES + r) * CH;
    out[rowoff + c0] = acc0[i] + bf[c0];
    out[rowoff + c1] = acc1[i] + bf[c1];
  }
}

// ---------------- Launcher ----------------
extern "C" void kernel_launch(void* const* d_in, const int* in_sizes, int n_in,
                              void* d_out, int out_size, void* d_ws, size_t ws_size,
                              hipStream_t stream) {
  (void)in_sizes; (void)n_in; (void)out_size; (void)ws_size;
  const float* x1d = (const float*)d_in[0];
  const float* lnw = (const float*)d_in[1];
  const float* lnb = (const float*)d_in[2];
  const float* wq  = (const float*)d_in[3];
  const float* wk  = (const float*)d_in[4];
  const float* wv  = (const float*)d_in[5];
  const float* wg  = (const float*)d_in[6];
  const float* bg  = (const float*)d_in[7];
  const float* wf  = (const float*)d_in[8];
  const float* bf  = (const float*)d_in[9];
  float* out = (float*)d_out;

  char* ws = (char*)d_ws;
  const size_t NBUF = (size_t)NPOS * CH * sizeof(_Float16);  // 33,554,432 B
  _Float16* wpk = (_Float16*)ws;                 // 655,360 B
  size_t off = 80ull * 8 * 32 * 16 * sizeof(_Float16);
  _Float16* qb = (_Float16*)(ws + off); off += NBUF;
  _Float16* kb = (_Float16*)(ws + off); off += NBUF;
  _Float16* vb = (_Float16*)(ws + off); off += NBUF;
  _Float16* gb = (_Float16*)(ws + off); off += NBUF;
  _Float16* ob = (_Float16*)(ws + off); off += NBUF;

  msa_pack_w<<<dim3(80), dim3(256), 0, stream>>>(wq, wk, wv, wg, wf, wpk);
  msa_ln_qkvg<<<dim3(NPOS / 16), dim3(256), 0, stream>>>(x1d, lnw, lnb, bg, wpk,
                                                         qb, kb, vb, gb);
  msa_attn<<<dim3((SEQ / 128) * RES * NH), dim3(256), LDS_TOTAL, stream>>>(qb, kb, vb, gb, ob);
  msa_outproj<<<dim3((SEQ / 16) * RES), dim3(256), 0, stream>>>(ob, wpk, bf, out);
}
